// _NonLocalBlockND_11355893530868
// MI455X (gfx1250) — matrix operand order repaired and checked by reading, not compiled, not run on hardware
//
#include <hip/hip_runtime.h>
#include <hip/hip_bf16.h>
#include <math.h>

// ---------------------------------------------------------------------------
// Problem constants (from the reference)
// ---------------------------------------------------------------------------
#define BB    4
#define CIN   128
#define CI    64
#define RED   32
#define HH    96
#define NPIX  (HH * HH)        // 9216
#define HP    48
#define MPIX  (HP * HP)        // 2304
#define BN_EPS 1e-5f

typedef __attribute__((ext_vector_type(16))) __bf16 v16bf;
typedef __attribute__((ext_vector_type(8)))  float  v8f;

// ---------------------------------------------------------------------------
// CDNA5 WMMA fragment layouts (bf16 16x16x32, f32 accumulate):
//   A (16x32): row = lane&15. lanes 0-15: V0..V3 <- K 0..7, V4..V7 <- K 16..23
//              lanes 16-31: K ranges offset by 8. => per lane, A is two
//              contiguous 8-element bf16 chunks at K = 8*hi and 16+8*hi.
//   B (32x16): col = lane&15, V0..V3 <- K 0..7, V4..V7 <- K 16..23 (+8 for lanes 16-31)
//              => per lane, B uses the same two 8-element K chunks as A.
//   C/D (16x16 f32): V[v] holds row v (+8 for lanes 16-31), col = lane&15.
// ---------------------------------------------------------------------------
__device__ __forceinline__ int a_kbase(int v, int hi) {
  return (v < 4 ? 2 * v : 16 + 2 * (v - 4)) + 8 * hi;
}

// A (16x32) from ROW-major bf16: elem(r,k) = base[r*ld + k]
__device__ __forceinline__ v16bf load_A_row_bf16(const __bf16* base, int ld) {
  int lane = threadIdx.x & 31, r = lane & 15, hi = lane >> 4;
  const __bf16* rp = base + r * ld + 8 * hi;
  v16bf a;
#pragma unroll
  for (int i = 0; i < 8; ++i) a[i] = rp[i];
#pragma unroll
  for (int i = 0; i < 8; ++i) a[8 + i] = rp[16 + i];
  return a;
}

// A (16x32) from K-major f32: elem(r,k) = base[k*ld + r]   (lane-coalesced)
__device__ __forceinline__ v16bf load_A_col_f32(const float* __restrict__ base, int ld) {
  int lane = threadIdx.x & 31, r = lane & 15, hi = lane >> 4;
  v16bf a;
#pragma unroll
  for (int v = 0; v < 8; ++v) {
    int k = a_kbase(v, hi);
    a[2 * v]     = (__bf16)base[k * ld + r];
    a[2 * v + 1] = (__bf16)base[(k + 1) * ld + r];
  }
  return a;
}

// B (32x16) from N-major bf16: elem(k,n) = base[n*ld + k]  (contig in k)
__device__ __forceinline__ v16bf load_B_n_bf16(const __bf16* base, int ld) {
  int lane = threadIdx.x & 31, n = lane & 15, hi = lane >> 4;
  const __bf16* rp = base + n * ld + 8 * hi;
  v16bf b;
#pragma unroll
  for (int i = 0; i < 16; ++i) b[i] = rp[i < 8 ? i : i + 8];
  return b;
}

// B (32x16) from N-major f32: elem(k,n) = base[n*ld + k]
__device__ __forceinline__ v16bf load_B_n_f32(const float* __restrict__ base, int ld) {
  int lane = threadIdx.x & 31, n = lane & 15, hi = lane >> 4;
  v16bf b;
#pragma unroll
  for (int v = 0; v < 8; ++v) {
    int k = a_kbase(v, hi);
    b[2 * v]     = (__bf16)base[n * ld + k];
    b[2 * v + 1] = (__bf16)base[n * ld + k + 1];
  }
  return b;
}

// A (16x32) from ROW-major f32: elem(r,k) = base[r*ld + k]
__device__ __forceinline__ v16bf load_A_row_f32(const float* __restrict__ base, int ld) {
  int lane = threadIdx.x & 31, r = lane & 15, hi = lane >> 4;
  v16bf a;
#pragma unroll
  for (int v = 0; v < 8; ++v) {
    int k = a_kbase(v, hi);
    a[2 * v]     = (__bf16)base[r * ld + k];
    a[2 * v + 1] = (__bf16)base[r * ld + k + 1];
  }
  return a;
}

__device__ __forceinline__ v8f wmma_bf16(v16bf a, v16bf b, v8f c) {
  return __builtin_amdgcn_wmma_f32_16x16x32_bf16(false, a, false, b, (short)0, c,
                                                 false, false);
}

__device__ __forceinline__ float sigmoidf_(float x) {
  return 1.f / (1.f + __expf(-x));
}

// ---------------------------------------------------------------------------
// Kernel 1: four fused 1x1 convs (K=128 -> 64 each) via WMMA.
// ---------------------------------------------------------------------------
__global__ __launch_bounds__(32) void conv4_kernel(
    const float* __restrict__ x,
    const float* __restrict__ w0, const float* __restrict__ bi0,
    const float* __restrict__ w1, const float* __restrict__ bi1,
    const float* __restrict__ w2, const float* __restrict__ bi2,
    const float* __restrict__ w3, const float* __restrict__ bi3,
    float* __restrict__ o0, float* __restrict__ o1,
    float* __restrict__ o2, float* __restrict__ o3) {
  int blk = blockIdx.x;
  int b = blk / (NPIX / 16);
  int n0 = (blk % (NPIX / 16)) * 16;
  const float* xb = x + (size_t)b * CIN * NPIX;
  int lane = threadIdx.x & 31, r = lane & 15, hi = lane >> 4;

  v16bf A[4];
#pragma unroll
  for (int kf = 0; kf < 4; ++kf)
    A[kf] = load_A_col_f32(xb + (size_t)kf * 32 * NPIX + n0, NPIX);

  const float* ws[4] = {w0, w1, w2, w3};
  const float* bs[4] = {bi0, bi1, bi2, bi3};
  float* os[4] = {o0, o1, o2, o3};

#pragma unroll
  for (int br = 0; br < 4; ++br) {
    const float* w = ws[br];
    float* out = os[br] + (size_t)b * CI * NPIX;
#pragma unroll
    for (int of = 0; of < 4; ++of) {
      v8f acc = {};
#pragma unroll
      for (int kf = 0; kf < 4; ++kf) {
        v16bf Bf = load_B_n_f32(w + (of * 16) * CIN + kf * 32, CIN);
        acc = wmma_bf16(A[kf], Bf, acc);
      }
      int oc = of * 16 + r;
      float bias = bs[br][oc];
#pragma unroll
      for (int v = 0; v < 8; ++v) {
        int n = n0 + v + 8 * hi;
        out[(size_t)oc * NPIX + n] = acc[v] + bias;
      }
    }
  }
}

// ---------------------------------------------------------------------------
// Kernel 2: 2x2 max pool
// ---------------------------------------------------------------------------
__global__ void pool2_kernel(const float* __restrict__ src, float* __restrict__ dst) {
  int i = blockIdx.x * blockDim.x + threadIdx.x;
  if (i >= BB * CI * HP * HP) return;
  int j = i % HP;
  int t = i / HP;
  int ii = t % HP;
  int bc = t / HP;
  const float* s = src + (size_t)bc * NPIX + (2 * ii) * HH + 2 * j;
  dst[i] = fmaxf(fmaxf(s[0], s[1]), fmaxf(s[HH], s[HH + 1]));
}

// ---------------------------------------------------------------------------
// Kernel 3: per-(b,c) spatial mean (SE squeeze)
// ---------------------------------------------------------------------------
__global__ __launch_bounds__(256) void mean_kernel(const float* __restrict__ src,
                                                   float* __restrict__ out, int S) {
  __shared__ float red[256];
  int bc = blockIdx.x;
  const float* s = src + (size_t)bc * S;
  float acc = 0.f;
  for (int i = threadIdx.x; i < S; i += 256) acc += s[i];
  red[threadIdx.x] = acc;
  __syncthreads();
  for (int st = 128; st > 0; st >>= 1) {
    if (threadIdx.x < st) red[threadIdx.x] += red[threadIdx.x + st];
    __syncthreads();
  }
  if (threadIdx.x == 0) out[bc] = red[0] / (float)S;
}

// ---------------------------------------------------------------------------
// Kernel 4: SE excite FCs -> sigmoid channel scales (0:g/se0 1:th/se1
// 2:phi/se3 3:thu/se3)
// ---------------------------------------------------------------------------
__global__ __launch_bounds__(64) void se_kernel(
    const float* __restrict__ means,
    const float* w1_0, const float* b1_0, const float* w2_0, const float* b2_0,
    const float* w1_1, const float* b1_1, const float* w2_1, const float* b2_1,
    const float* w1_3, const float* b1_3, const float* w2_3, const float* b2_3,
    float* __restrict__ scales) {
  __shared__ float y1[RED];
  __shared__ float mv[CI];
  int t = blockIdx.x >> 2;
  int b = blockIdx.x & 3;
  const float *w1, *bb1, *w2, *bb2;
  if (t == 0)      { w1 = w1_0; bb1 = b1_0; w2 = w2_0; bb2 = b2_0; }
  else if (t == 1) { w1 = w1_1; bb1 = b1_1; w2 = w2_1; bb2 = b2_1; }
  else             { w1 = w1_3; bb1 = b1_3; w2 = w2_3; bb2 = b2_3; }
  int tid = threadIdx.x;
  mv[tid] = means[(t * BB + b) * CI + tid];
  __syncthreads();
  if (tid < RED) {
    float a = bb1[tid];
    for (int c = 0; c < CI; ++c) a += w1[tid * CI + c] * mv[c];
    y1[tid] = fmaxf(a, 0.f);
  }
  __syncthreads();
  float a = bb2[tid];
  for (int j = 0; j < RED; ++j) a += w2[tid * RED + j] * y1[j];
  scales[(t * BB + b) * CI + tid] = sigmoidf_(a);
}

// ---------------------------------------------------------------------------
// Kernel 5: fold per-channel scale in place (c-major (B,64,S))
// ---------------------------------------------------------------------------
__global__ void scale_kernel(const float* __restrict__ src,
                             const float* __restrict__ scale,
                             float* __restrict__ dst, int S) {
  int i = blockIdx.x * blockDim.x + threadIdx.x;
  if (i >= BB * CI * S) return;
  dst[i] = src[i] * scale[i / S];
}

// ---------------------------------------------------------------------------
// Kernel 6: ConvGRU via WMMA. One wave = 16 pixels.
//   z/r: K=128 GEMM over [x | h]; n_x: K=64 GEMM over x   (40 WMMAs)
//   gates in C layout -> r*h -> bf16 LDS transpose -> A layout
//   n_h: K=64 GEMM over r*h                              ( 8 WMMAs)
// xin must already carry the SE scale.
// ---------------------------------------------------------------------------
__global__ __launch_bounds__(32) void gru_wmma_kernel(
    const float* __restrict__ xin, const float* __restrict__ hin,
    const float* __restrict__ xz_w, const float* __restrict__ xz_b,
    const float* __restrict__ xr_w, const float* __restrict__ xr_b,
    const float* __restrict__ xn_w, const float* __restrict__ xn_b,
    const float* __restrict__ hz_w, const float* __restrict__ hz_b,
    const float* __restrict__ hr_w, const float* __restrict__ hr_b,
    const float* __restrict__ hn_w, const float* __restrict__ hn_b,
    float* __restrict__ out, int S) {
  __shared__ __bf16 lds_rh[16][72];  // 16 px x 64 c, padded row (144B, 16B-mult)
  int blk = blockIdx.x;
  int b = blk / (S / 16);
  int n0 = (blk % (S / 16)) * 16;
  const float* xb = xin + (size_t)b * CI * S;
  const float* hb = hin + (size_t)b * CI * S;
  int lane = threadIdx.x & 31, r = lane & 15, hi = lane >> 4;

  v16bf Ag[2], Ah[2];
#pragma unroll
  for (int kf = 0; kf < 2; ++kf) {
    Ag[kf] = load_A_col_f32(xb + (size_t)kf * 32 * S + n0, S);
    Ah[kf] = load_A_col_f32(hb + (size_t)kf * 32 * S + n0, S);
  }

  v8f zacc[4], nacc[4];
  float hv[4][8];
#pragma unroll
  for (int nf = 0; nf < 4; ++nf) {
    v8f za = {}, ra = {}, nx = {};
#pragma unroll
    for (int kf = 0; kf < 2; ++kf) {
      za = wmma_bf16(Ag[kf], load_B_n_f32(xz_w + (nf * 16) * CI + kf * 32, CI), za);
      za = wmma_bf16(Ah[kf], load_B_n_f32(hz_w + (nf * 16) * CI + kf * 32, CI), za);
      ra = wmma_bf16(Ag[kf], load_B_n_f32(xr_w + (nf * 16) * CI + kf * 32, CI), ra);
      ra = wmma_bf16(Ah[kf], load_B_n_f32(hr_w + (nf * 16) * CI + kf * 32, CI), ra);
      nx = wmma_bf16(Ag[kf], load_B_n_f32(xn_w + (nf * 16) * CI + kf * 32, CI), nx);
    }
    int o = nf * 16 + r;
    float zb = xz_b[o] + hz_b[o];
    float rb = xr_b[o] + hr_b[o];
    const float* hrow = hb + (size_t)o * S + n0;
#pragma unroll
    for (int v = 0; v < 8; ++v) {
      float h = hrow[v + 8 * hi];
      hv[nf][v] = h;
      float zv = sigmoidf_(za[v] + zb);
      float rv = sigmoidf_(ra[v] + rb);
      za[v] = zv;                               // keep z in zacc
      lds_rh[v + 8 * hi][o] = (__bf16)(rv * h); // r*h tile for transpose
    }
    zacc[nf] = za;
    nacc[nf] = nx;
  }
  __syncthreads();

  v16bf Arh[2];
#pragma unroll
  for (int kf = 0; kf < 2; ++kf)
    Arh[kf] = load_A_row_bf16(&lds_rh[0][0] + kf * 32, 72);

#pragma unroll
  for (int nf = 0; nf < 4; ++nf) {
    v8f na = nacc[nf];
#pragma unroll
    for (int kf = 0; kf < 2; ++kf)
      na = wmma_bf16(Arh[kf], load_B_n_f32(hn_w + (nf * 16) * CI + kf * 32, CI), na);
    int o = nf * 16 + r;
    float nb = xn_b[o] + hn_b[o];
    float* orow = out + (size_t)b * CI * S + (size_t)o * S + n0;
#pragma unroll
    for (int v = 0; v < 8; ++v) {
      float nn = tanhf(na[v] + nb);
      float z = zacc[nf][v];
      float hnew = (1.f - z) * hv[nf][v] + z * nn;
      orow[v + 8 * hi] = hnew >= 0.f ? hnew : 0.2f * hnew;
    }
  }
}

// ---------------------------------------------------------------------------
// Kernel 7a/7b/7c: bf16 packing for the attention operands.
//   theta -> (b,n,c) row-major; phi*scale -> (b,m,c) row-major; g -> (b,c,m)
// ---------------------------------------------------------------------------
__global__ void pack_rowmajor_bf16(const float* __restrict__ src,  // (B,64,S)
                                   __bf16* __restrict__ dst,       // (B,S,64)
                                   int S) {
  int i = blockIdx.x * blockDim.x + threadIdx.x;
  if (i >= BB * CI * S) return;
  int c = i & 63;
  int t = i >> 6;
  int n = t % S;
  int b = t / S;
  dst[i] = (__bf16)src[(size_t)b * CI * S + (size_t)c * S + n];
}

__global__ void pack_rowmajor_scaled_bf16(const float* __restrict__ src,
                                          const float* __restrict__ scale,
                                          __bf16* __restrict__ dst, int S) {
  int i = blockIdx.x * blockDim.x + threadIdx.x;
  if (i >= BB * CI * S) return;
  int c = i & 63;
  int t = i >> 6;
  int n = t % S;
  int b = t / S;
  dst[i] = (__bf16)(src[(size_t)b * CI * S + (size_t)c * S + n] * scale[b * CI + c]);
}

__global__ void pack_cmajor_bf16(const float* __restrict__ src,
                                 __bf16* __restrict__ dst, int tot) {
  int i = blockIdx.x * blockDim.x + threadIdx.x;
  if (i < tot) dst[i] = (__bf16)src[i];
}

// ---------------------------------------------------------------------------
// Kernel 8: flash attention, all-bf16 operands, fp32 accumulate.
// ---------------------------------------------------------------------------
__global__ __launch_bounds__(32) void attn_kernel(
    const __bf16* __restrict__ thA,   // (B,NPIX,64) row-major
    const __bf16* __restrict__ phiT,  // (B,MPIX,64) row-major, SE-scaled
    const __bf16* __restrict__ gB,    // (B,64,MPIX) c-major
    const float* __restrict__ thu,    // (B,64,NPIX) pre-SE
    const float* __restrict__ s_thu,  // [B][64]
    float* __restrict__ y_buf) {      // (B,NPIX,64) row-major
  __shared__ __bf16 lds_p[16][48];    // 16 x 32 P tile (96B rows)
  int blk = blockIdx.x;
  int b = blk / (NPIX / 16);
  int n0 = (blk % (NPIX / 16)) * 16;
  const __bf16* thb = thA + (size_t)b * NPIX * CI;
  const __bf16* phb = phiT + (size_t)b * MPIX * CI;
  const __bf16* gb = gB + (size_t)b * CI * MPIX;
  int lane = threadIdx.x & 31, r = lane & 15, hi = lane >> 4;

  v16bf A0 = load_A_row_bf16(thb + (size_t)n0 * CI, CI);
  v16bf A1 = load_A_row_bf16(thb + (size_t)n0 * CI + 32, CI);

  v8f O0 = {}, O1 = {}, O2 = {}, O3 = {};
  float mrow[8], lrow[8];
#pragma unroll
  for (int v = 0; v < 8; ++v) { mrow[v] = -3.0e38f; lrow[v] = 0.f; }

  for (int m0 = 0; m0 < MPIX; m0 += 32) {
    // scores: B = phi tile (K=c, N=m) from (m,c) row-major
    const __bf16* pb = phb + (size_t)m0 * CI;
    v16bf Bp00 = load_B_n_bf16(pb, CI);
    v16bf Bp10 = load_B_n_bf16(pb + 32, CI);
    v16bf Bp01 = load_B_n_bf16(pb + 16 * CI, CI);
    v16bf Bp11 = load_B_n_bf16(pb + 16 * CI + 32, CI);
    v8f S0 = {}, S1 = {};
    S0 = wmma_bf16(A0, Bp00, S0);
    S0 = wmma_bf16(A1, Bp10, S0);
    S1 = wmma_bf16(A0, Bp01, S1);
    S1 = wmma_bf16(A1, Bp11, S1);

    float alpha[8];
#pragma unroll
    for (int v = 0; v < 8; ++v) {
      float t = fmaxf(S0[v], S1[v]);
#pragma unroll
      for (int msk = 1; msk <= 8; msk <<= 1) t = fmaxf(t, __shfl_xor(t, msk, 32));
      float mn = fmaxf(mrow[v], t);
      alpha[v] = __expf(mrow[v] - mn);
      mrow[v] = mn;
      float p0 = __expf(S0[v] - mn);
      float p1 = __expf(S1[v] - mn);
      float s = p0 + p1;
#pragma unroll
      for (int msk = 1; msk <= 8; msk <<= 1) s += __shfl_xor(s, msk, 32);
      lrow[v] = lrow[v] * alpha[v] + s;
      S0[v] = p0;
      S1[v] = p1;
    }
#pragma unroll
    for (int v = 0; v < 8; ++v) {
      O0[v] *= alpha[v]; O1[v] *= alpha[v];
      O2[v] *= alpha[v]; O3[v] *= alpha[v];
    }

    // P: C layout -> bf16 LDS -> A layout
#pragma unroll
    for (int v = 0; v < 8; ++v) {
      lds_p[v + 8 * hi][r] = (__bf16)S0[v];
      lds_p[v + 8 * hi][16 + r] = (__bf16)S1[v];
    }
    __syncthreads();
    v16bf Ap = load_A_row_bf16(&lds_p[0][0], 48);
    __syncthreads();

    // P @ G: B = g tile (K=m, N=c) from (c,m) storage (contig in m)
    v16bf Bg0 = load_B_n_bf16(gb + m0, MPIX);
    v16bf Bg1 = load_B_n_bf16(gb + (size_t)16 * MPIX + m0, MPIX);
    v16bf Bg2 = load_B_n_bf16(gb + (size_t)32 * MPIX + m0, MPIX);
    v16bf Bg3 = load_B_n_bf16(gb + (size_t)48 * MPIX + m0, MPIX);
    O0 = wmma_bf16(Ap, Bg0, O0);
    O1 = wmma_bf16(Ap, Bg1, O1);
    O2 = wmma_bf16(Ap, Bg2, O2);
    O3 = wmma_bf16(Ap, Bg3, O3);
  }

  const float* thub = thu + (size_t)b * CI * NPIX;
  float* yb = y_buf + (size_t)b * NPIX * CI;
  v8f Os[4] = {O0, O1, O2, O3};
#pragma unroll
  for (int j = 0; j < 4; ++j) {
    int c = j * 16 + r;
    float sc = s_thu[b * CI + c];
#pragma unroll
    for (int v = 0; v < 8; ++v) {
      int n = n0 + v + 8 * hi;
      yb[(size_t)n * CI + c] = Os[j][v] / lrow[v] + thub[(size_t)c * NPIX + n] * sc;
    }
  }
}

// ---------------------------------------------------------------------------
// Kernel 9: output 1x1 conv (64 -> 128) + BatchNorm + residual, WMMA
// ---------------------------------------------------------------------------
__global__ __launch_bounds__(32) void wconv_kernel(
    const float* __restrict__ y_buf, const float* __restrict__ x,
    const float* __restrict__ Ww, const float* __restrict__ Wb,
    const float* __restrict__ gamma, const float* __restrict__ beta,
    const float* __restrict__ mean, const float* __restrict__ var,
    float* __restrict__ z) {
  int blk = blockIdx.x;
  int b = blk / (NPIX / 16);
  int n0 = (blk % (NPIX / 16)) * 16;
  const float* yb = y_buf + (size_t)b * NPIX * CI + (size_t)n0 * CI;
  int lane = threadIdx.x & 31, r = lane & 15, hi = lane >> 4;

  v16bf A0 = load_A_row_f32(yb, CI);
  v16bf A1 = load_A_row_f32(yb + 32, CI);

#pragma unroll
  for (int of = 0; of < 8; ++of) {
    v16bf B0 = load_B_n_f32(Ww + (of * 16) * CI, CI);
    v16bf B1 = load_B_n_f32(Ww + (of * 16) * CI + 32, CI);
    v8f acc = {};
    acc = wmma_bf16(A0, B0, acc);
    acc = wmma_bf16(A1, B1, acc);
    int o = of * 16 + r;
    float bias = Wb[o];
    float rs = rsqrtf(var[o] + BN_EPS) * gamma[o];
    float mu = mean[o];
    float be = beta[o];
    const float* xr = x + (size_t)b * CIN * NPIX + (size_t)o * NPIX;
    float* zr = z + (size_t)b * CIN * NPIX + (size_t)o * NPIX;
#pragma unroll
    for (int v = 0; v < 8; ++v) {
      int n = n0 + v + 8 * hi;
      zr[n] = (acc[v] + bias - mu) * rs + be + xr[n];
    }
  }
}

// ---------------------------------------------------------------------------
// Host-side launch. d_in order = setup_inputs() insertion order, params
// flattened depth-first (see round-0 notes).
// d_out = [ z | ret_corr1 | ret_corr2 ]
// ---------------------------------------------------------------------------
extern "C" void kernel_launch(void* const* d_in, const int* in_sizes, int n_in,
                              void* d_out, int out_size, void* d_ws, size_t ws_size,
                              hipStream_t stream) {
  const float* x = (const float*)d_in[0];
  const float* corr1 = (const float*)d_in[1];
  const float* corr2 = (const float*)d_in[2];
  const float* g_w = (const float*)d_in[3];
  const float* g_b = (const float*)d_in[4];
  const float* th_w = (const float*)d_in[5];
  const float* th_b = (const float*)d_in[6];
  const float* ph_w = (const float*)d_in[7];
  const float* ph_b = (const float*)d_in[8];
  const float* tu_w = (const float*)d_in[9];
  const float* tu_b = (const float*)d_in[10];
  const float* se0_w1 = (const float*)d_in[11];
  const float* se0_b1 = (const float*)d_in[12];
  const float* se0_w2 = (const float*)d_in[13];
  const float* se0_b2 = (const float*)d_in[14];
  const float* se1_w1 = (const float*)d_in[15];
  const float* se1_b1 = (const float*)d_in[16];
  const float* se1_w2 = (const float*)d_in[17];
  const float* se1_b2 = (const float*)d_in[18];
  const float* se3_w1 = (const float*)d_in[19];
  const float* se3_b1 = (const float*)d_in[20];
  const float* se3_w2 = (const float*)d_in[21];
  const float* se3_b2 = (const float*)d_in[22];
  const float* g1[12];
  const float* g2[12];
  for (int i = 0; i < 12; ++i) {
    g1[i] = (const float*)d_in[23 + i];
    g2[i] = (const float*)d_in[35 + i];
  }
  const float* W_w = (const float*)d_in[47];
  const float* W_b = (const float*)d_in[48];
  const float* bn_g = (const float*)d_in[49];
  const float* bn_b = (const float*)d_in[50];
  const float* bn_m = (const float*)d_in[51];
  const float* bn_v = (const float*)d_in[52];

  float* out = (float*)d_out;
  float* z_out = out;                          // B*128*NPIX
  float* rc1 = out + (size_t)BB * CIN * NPIX;  // B*64*MPIX  (= g_x)
  float* rc2 = rc1 + (size_t)BB * CI * MPIX;   // B*64*NPIX  (= theta_x)

  // workspace layout (floats)
  float* ws = (float*)d_ws;
  const size_t FULL = (size_t)BB * CI * NPIX;  // 2,359,296
  const size_t HALF = (size_t)BB * CI * MPIX;  //   589,824
  float* th_pre = ws;                   // theta conv (SE scale folded in later)
  float* tu_pre = th_pre + FULL;        // theta_u conv (pre-SE, kept raw)
  float* g_full = tu_pre + FULL;        // g conv full-res; later reused: y_buf
  float* ph_full = g_full + FULL;       // phi conv full-res; later: bf16 packs
  float* g_pre = ph_full + FULL;        // pooled g (SE scale folded in later)
  float* ph_pre = g_pre + HALF;         // pooled phi (raw)
  float* means = ph_pre + HALF;         // [4][B][64]
  float* scales = means + 4 * BB * CI;  // [4][B][64]
  float* y_buf = g_full;                // alias (g_full dead after pooling)
  // bf16 packs carved out of ph_full (dead after pooling): 7.08MB < 9.44MB
  __bf16* thA = (__bf16*)ph_full;                 // (B,NPIX,64)
  __bf16* phiT = thA + (size_t)BB * NPIX * CI;    // (B,MPIX,64)
  __bf16* gBp = phiT + (size_t)BB * MPIX * CI;    // (B,64,MPIX)

  const int tiles = BB * NPIX / 16;  // 2304

  // 1) four branch convs
  conv4_kernel<<<tiles, 32, 0, stream>>>(x, g_w, g_b, th_w, th_b, ph_w, ph_b,
                                         tu_w, tu_b, g_full, th_pre, ph_full,
                                         tu_pre);
  // 2) pooling
  {
    int tot = BB * CI * MPIX;
    pool2_kernel<<<(tot + 255) / 256, 256, 0, stream>>>(g_full, g_pre);
    pool2_kernel<<<(tot + 255) / 256, 256, 0, stream>>>(ph_full, ph_pre);
  }
  // 3) SE means (g, theta, phi, theta_u)
  mean_kernel<<<BB * CI, 256, 0, stream>>>(g_pre, means + 0 * BB * CI, MPIX);
  mean_kernel<<<BB * CI, 256, 0, stream>>>(th_pre, means + 1 * BB * CI, NPIX);
  mean_kernel<<<BB * CI, 256, 0, stream>>>(ph_pre, means + 2 * BB * CI, MPIX);
  mean_kernel<<<BB * CI, 256, 0, stream>>>(tu_pre, means + 3 * BB * CI, NPIX);
  // 4) SE excitation -> scales
  se_kernel<<<16, 64, 0, stream>>>(means, se0_w1, se0_b1, se0_w2, se0_b2,
                                   se1_w1, se1_b1, se1_w2, se1_b2, se3_w1,
                                   se3_b1, se3_w2, se3_b2, scales);
  // 5) fold SE scales into the GRU inputs (in place; raw values dead here)
  scale_kernel<<<(int)((HALF + 255) / 256), 256, 0, stream>>>(
      g_pre, scales + 0 * BB * CI, g_pre, MPIX);
  scale_kernel<<<(int)((FULL + 255) / 256), 256, 0, stream>>>(
      th_pre, scales + 1 * BB * CI, th_pre, NPIX);
  // 6) ConvGRUs via WMMA (outputs written straight into d_out)
  gru_wmma_kernel<<<BB * MPIX / 16, 32, 0, stream>>>(
      g_pre, corr1, g1[0], g1[1], g1[2], g1[3], g1[4], g1[5], g1[6], g1[7],
      g1[8], g1[9], g1[10], g1[11], rc1, MPIX);
  gru_wmma_kernel<<<BB * NPIX / 16, 32, 0, stream>>>(
      th_pre, corr2, g2[0], g2[1], g2[2], g2[3], g2[4], g2[5], g2[6], g2[7],
      g2[8], g2[9], g2[10], g2[11], rc2, NPIX);
  // 7) bf16 packs for attention
  pack_rowmajor_bf16<<<(int)((FULL + 255) / 256), 256, 0, stream>>>(rc2, thA,
                                                                    NPIX);
  pack_rowmajor_scaled_bf16<<<(int)((HALF + 255) / 256), 256, 0, stream>>>(
      ph_pre, scales + 2 * BB * CI, phiT, MPIX);
  pack_cmajor_bf16<<<(int)((HALF + 255) / 256), 256, 0, stream>>>(
      rc1, gBp, (int)HALF);
  // 8) flash attention (+ SE-scaled theta_u residual)
  attn_kernel<<<tiles, 32, 0, stream>>>(thA, phiT, gBp, tu_pre,
                                        scales + 3 * BB * CI, y_buf);
  // 9) output conv + BN + residual
  wconv_kernel<<<tiles, 32, 0, stream>>>(y_buf, x, W_w, W_b, bn_g, bn_b, bn_m,
                                         bn_v, z_out);
}
